// RoPESelfKANtentionND_85323820302821
// MI455X (gfx1250) — matrix operand order repaired and checked by reading, not compiled, not run on hardware
//
#include <hip/hip_runtime.h>
#include <hip/hip_bf16.h>
#include <stdint.h>

typedef __attribute__((ext_vector_type(16))) __bf16 v16bf;
typedef __attribute__((ext_vector_type(2)))  __bf16 v2bf;
typedef __attribute__((ext_vector_type(8)))  float  v8f;

#define BB 8
#define CC 64
#define HH 50
#define WW 65
#define NN 3250
#define NP 3264      // padded to 204*16
#define MT 204       // m tiles of 16
#define NT 102       // n tiles of 32
#define K2 576       // C*3*3

static __device__ __forceinline__ v8f wmma_bf16(v16bf a, v16bf b, v8f c) {
  return __builtin_amdgcn_wmma_f32_16x16x32_bf16(false, a, false, b, (short)0, c, false, false);
}

// async copy: 16B per lane, global -> LDS, tracked by ASYNCcnt
static __device__ __forceinline__ void async_b128(uint32_t lds_addr, const void* gaddr) {
  asm volatile("global_load_async_to_lds_b128 %0, %1, off"
               :: "v"(lds_addr), "v"(gaddr) : "memory");
}

// ---------------- weight fp32 -> bf16 ----------------
__global__ void cvt_w_kernel(const float* __restrict__ wq, const float* __restrict__ wk,
                             const float* __restrict__ wv,
                             __bf16* __restrict__ wqb, __bf16* __restrict__ wkb,
                             __bf16* __restrict__ wvb) {
  int i = blockIdx.x * blockDim.x + threadIdx.x;
  if (i >= CC * K2) return;
  wqb[i] = (__bf16)wq[i];
  wkb[i] = (__bf16)wk[i];
  wvb[i] = (__bf16)wv[i];
}

// ---------------- zero the pad rows ----------------
__global__ void pad_zero_kernel(__bf16* __restrict__ qr, __bf16* __restrict__ kr,
                                __bf16* __restrict__ vT) {
  int i = blockIdx.x * blockDim.x + threadIdx.x;
  if (i >= BB * (NP - NN) * CC) return;
  int b = i / ((NP - NN) * CC);
  int rr = (i / CC) % (NP - NN);
  int c = i % CC;
  int row = NN + rr;
  qr[(b * NP + row) * CC + c] = (__bf16)0.f;
  kr[(b * NP + row) * CC + c] = (__bf16)0.f;
  vT[(b * CC + c) * NP + row] = (__bf16)0.f;
}

// ---------------- qkv conv (implicit GEMM, WMMA bf16) + bias + RoPE ----------------
__global__ __launch_bounds__(96) void conv_qkv_kernel(
    const float* __restrict__ x,
    const __bf16* __restrict__ wqb, const __bf16* __restrict__ wkb,
    const __bf16* __restrict__ wvb,
    const float* __restrict__ bq, const float* __restrict__ bk,
    const float* __restrict__ bv,
    const float* __restrict__ freqs,
    __bf16* __restrict__ qr, __bf16* __restrict__ kr, __bf16* __restrict__ vT) {
  __shared__ __align__(16) __bf16 Acol[16 * K2];   // im2col tile: [m][k], k=ci*9+kh*3+kw
  const int blk = blockIdx.x;
  const int b = blk / MT, mt = blk % MT;
  const int row0 = mt * 16;
  const int tid = threadIdx.x;

  // cooperative im2col build (SAME padding)
  for (int e = tid; e < 16 * K2; e += 96) {
    int m = e / K2, k = e % K2;
    int ci = k / 9, rr = k % 9, kh = rr / 3, kw = rr % 3;
    int p = row0 + m; if (p > NN - 1) p = NN - 1;
    int h = p / WW, w = p % WW;
    int hh = h + kh - 1, ww = w + kw - 1;
    float val = 0.f;
    if (hh >= 0 && hh < HH && ww >= 0 && ww < WW)
      val = x[(b * CC + ci) * NN + hh * WW + ww];
    Acol[e] = (__bf16)val;
  }
  __syncthreads();

  const int wave = tid >> 5;
  const int lane = tid & 31;
  const int lo = lane & 15, hi = lane >> 4;
  const __bf16* wb = (wave == 0) ? wqb : (wave == 1) ? wkb : wvb;

  v8f acc[4];
  #pragma unroll
  for (int ct = 0; ct < 4; ++ct) acc[ct] = (v8f){0, 0, 0, 0, 0, 0, 0, 0};

  for (int ks = 0; ks < K2 / 32; ++ks) {
    v16bf af;
    #pragma unroll
    for (int j = 0; j < 8; ++j) {
      int kp = ks * 32 + ((j < 4) ? 0 : 16) + hi * 8 + (j & 3) * 2;
      v2bf pr = *reinterpret_cast<const v2bf*>(&Acol[lo * K2 + kp]);
      af[2 * j] = pr.x; af[2 * j + 1] = pr.y;
    }
    #pragma unroll
    for (int ct = 0; ct < 4; ++ct) {
      v16bf bfm;
      #pragma unroll
      for (int j = 0; j < 8; ++j) {
        int kp = ks * 32 + ((j < 4) ? 0 : 16) + hi * 8 + (j & 3) * 2;
        v2bf pr = *reinterpret_cast<const v2bf*>(&wb[(ct * 16 + lo) * K2 + kp]);
        bfm[2 * j] = pr.x; bfm[2 * j + 1] = pr.y;
      }
      acc[ct] = wmma_bf16(af, bfm, acc[ct]);
    }
  }

  const float* bias = (wave == 0) ? bq : (wave == 1) ? bk : bv;
  #pragma unroll
  for (int ct = 0; ct < 4; ++ct) {
    #pragma unroll
    for (int r = 0; r < 8; ++r) {
      int c = ct * 16 + lo;
      int m = r + 8 * hi;
      int p = row0 + m;
      float val = acc[ct][r] + bias[c];
      if (wave < 2) {                 // wave-uniform branch: shfl is safe
        float part = __shfl_xor(val, 1);
        float cs = 1.f, sn = 0.f;
        if (p >= 1 && p < NN) {
          float t = (float)(p - 1);
          float ty = floorf(t / 57.0f);
          float tx = t - 57.0f * ty;
          int c2 = c >> 1;
          float ang = tx * freqs[c2] + ty * freqs[32 + c2];
          __sincosf(ang, &sn, &cs);
        }
        float res = (c & 1) ? (val * cs + part * sn) : (val * cs - part * sn);
        if (p < NN) {
          __bf16* dst = (wave == 0) ? qr : kr;
          dst[(b * NP + p) * CC + c] = (__bf16)res;
        }
      } else {
        if (p < NN) vT[(b * CC + c) * NP + p] = (__bf16)val;
      }
    }
  }
}

// ---------------- flash attention, 4 waves/block, async double-buffered K/V ----------------
__global__ __launch_bounds__(128) void attn_kernel(
    const __bf16* __restrict__ qr, const __bf16* __restrict__ kr,
    const __bf16* __restrict__ vT, float* __restrict__ attout) {
  __shared__ __align__(16) __bf16 Kt[2][32 * 64];   // [n_local][c]
  __shared__ __align__(16) __bf16 Vt[2][64 * 32];   // [c][n_local]
  __shared__ __align__(16) __bf16 Plds[4][16 * 32];

  const int blk = blockIdx.x;
  const int b = blk / 51;
  const int tid = threadIdx.x;
  const int wave = tid >> 5;
  const int lane = tid & 31;
  const int mt = (blk % 51) * 4 + wave;              // 204 tiles / block-group of 4 waves
  const int row0 = mt * 16;
  const int lo = lane & 15, hi = lane >> 4;

  // stage one 32-key tile of K and V into LDS buffer `buf` (4 async b128 per wave)
  auto stage = [&](int buf, int n0s) {
    const char* kbase = (const char*)kr + (size_t)(b * NP + n0s) * CC * 2;
    uint32_t klds = (uint32_t)(size_t)(&Kt[buf][0]);
    uint32_t vlds = (uint32_t)(size_t)(&Vt[buf][0]);
    #pragma unroll
    for (int i = 0; i < 2; ++i) {
      int chunk = i * 128 + tid;                     // 0..255 chunks of 16B
      async_b128(klds + chunk * 16, kbase + chunk * 16);   // K tile is contiguous 4KB
      int c = chunk >> 2, of = (chunk & 3) * 16;           // V: 64B row per channel
      const char* vsrc = (const char*)vT + (size_t)((b * CC + c) * NP + n0s) * 2 + of;
      async_b128(vlds + chunk * 16, vsrc);
    }
  };

  // Q tile A-fragments: 16 rows x 64 channels -> two 16x32 fragments
  v16bf qA[2];
  #pragma unroll
  for (int s = 0; s < 2; ++s) {
    #pragma unroll
    for (int j = 0; j < 8; ++j) {
      int kp = s * 32 + ((j < 4) ? 0 : 16) + hi * 8 + (j & 3) * 2;
      v2bf pr = *reinterpret_cast<const v2bf*>(&qr[(b * NP + row0 + lo) * CC + kp]);
      qA[s][2 * j] = pr.x; qA[s][2 * j + 1] = pr.y;
    }
  }

  v8f O[4];
  #pragma unroll
  for (int ct = 0; ct < 4; ++ct) O[ct] = (v8f){0, 0, 0, 0, 0, 0, 0, 0};
  float rm[8], rl[8];
  #pragma unroll
  for (int r = 0; r < 8; ++r) { rm[r] = -1e30f; rl[r] = 0.f; }

  stage(0, 0);                                        // prologue: tile 0 -> buffer 0

  for (int nt = 0; nt < NT; ++nt) {
    const int cur = nt & 1;
    const int n0 = nt * 32;
    const int n0n = (nt + 1 < NT) ? (n0 + 32) : n0;   // clamped; extra tile never consumed
    stage(cur ^ 1, n0n);                              // prefetch next tile
    asm volatile("s_wait_asynccnt 0x4" ::: "memory"); // tile nt's copies are done
    __syncthreads();

    const __bf16* Kc = &Kt[cur][0];
    const __bf16* Vc = &Vt[cur][0];

    v8f e0 = (v8f){0, 0, 0, 0, 0, 0, 0, 0};
    v8f e1 = (v8f){0, 0, 0, 0, 0, 0, 0, 0};
    #pragma unroll
    for (int s = 0; s < 2; ++s) {
      v16bf kB0, kB1;
      #pragma unroll
      for (int j = 0; j < 8; ++j) {
        int kp = s * 32 + ((j < 4) ? 0 : 16) + hi * 8 + (j & 3) * 2;
        v2bf p0 = *reinterpret_cast<const v2bf*>(&Kc[lo * CC + kp]);
        v2bf p1 = *reinterpret_cast<const v2bf*>(&Kc[(16 + lo) * CC + kp]);
        kB0[2 * j] = p0.x; kB0[2 * j + 1] = p0.y;
        kB1[2 * j] = p1.x; kB1[2 * j + 1] = p1.y;
      }
      e0 = wmma_bf16(qA[s], kB0, e0);
      e1 = wmma_bf16(qA[s], kB1, e1);
    }

    // mask key-tail columns
    if (n0 + lo >= NN) {
      #pragma unroll
      for (int r = 0; r < 8; ++r) e0[r] = -1e30f;
    }
    if (n0 + 16 + lo >= NN) {
      #pragma unroll
      for (int r = 0; r < 8; ++r) e1[r] = -1e30f;
    }

    // online softmax per row (rows live in 16-lane halves; xor<16 stays in-half)
    float scale[8];
    #pragma unroll
    for (int r = 0; r < 8; ++r) {
      float mx = fmaxf(e0[r], e1[r]);
      #pragma unroll
      for (int d = 1; d < 16; d <<= 1) mx = fmaxf(mx, __shfl_xor(mx, d));
      float nm = fmaxf(rm[r], mx);
      float sc = __expf(rm[r] - nm);
      rm[r] = nm;
      float p0 = __expf(e0[r] - nm);
      float p1 = __expf(e1[r] - nm);
      float s2 = p0 + p1;
      #pragma unroll
      for (int d = 1; d < 16; d <<= 1) s2 += __shfl_xor(s2, d);
      rl[r] = rl[r] * sc + s2;
      scale[r] = sc;
      int m = r + 8 * hi;
      Plds[wave][m * 32 + lo] = (__bf16)p0;           // D-layout -> A-layout via LDS
      Plds[wave][m * 32 + 16 + lo] = (__bf16)p1;
    }
    #pragma unroll
    for (int ct = 0; ct < 4; ++ct)
      #pragma unroll
      for (int r = 0; r < 8; ++r) O[ct][r] *= scale[r];

    // P as 16x32 A-fragment (same-wave LDS is in-order; no barrier needed)
    v16bf PA;
    #pragma unroll
    for (int j = 0; j < 8; ++j) {
      int kp = ((j < 4) ? 0 : 16) + hi * 8 + (j & 3) * 2;
      v2bf pr = *reinterpret_cast<const v2bf*>(&Plds[wave][lo * 32 + kp]);
      PA[2 * j] = pr.x; PA[2 * j + 1] = pr.y;
    }
    #pragma unroll
    for (int ct = 0; ct < 4; ++ct) {
      v16bf vB;
      #pragma unroll
      for (int j = 0; j < 8; ++j) {
        int kp = ((j < 4) ? 0 : 16) + hi * 8 + (j & 3) * 2;
        v2bf pr = *reinterpret_cast<const v2bf*>(&Vc[(ct * 16 + lo) * 32 + kp]);
        vB[2 * j] = pr.x; vB[2 * j + 1] = pr.y;
      }
      O[ct] = wmma_bf16(PA, vB, O[ct]);
    }
    __syncthreads();   // all waves done with buffer `cur` before it is re-staged
  }

  #pragma unroll
  for (int ct = 0; ct < 4; ++ct) {
    #pragma unroll
    for (int r = 0; r < 8; ++r) {
      int m = r + 8 * hi;
      int p = row0 + m;
      if (p < NN) {
        int c = ct * 16 + lo;
        attout[(b * CC + c) * NN + p] = O[ct][r] / rl[r];
      }
    }
  }
}

// ---------------- batchnorm reduce: per-channel mean / inv-std of y = g*x + out ----------------
__global__ __launch_bounds__(256) void bn_reduce_kernel(
    const float* __restrict__ x, const float* __restrict__ attout,
    const float* __restrict__ gamma, float* __restrict__ meanbuf,
    float* __restrict__ istdbuf) {
  __shared__ float ssum[256], ssq[256];
  const int c = blockIdx.x;
  const int tid = threadIdx.x;
  const float g = gamma[0];
  float s = 0.f, q = 0.f;
  for (int i = tid; i < BB * NN; i += 256) {
    int b = i / NN, p = i % NN;
    float y = g * x[(b * CC + c) * NN + p] + attout[(b * CC + c) * NN + p];
    s += y; q += y * y;
  }
  ssum[tid] = s; ssq[tid] = q;
  __syncthreads();
  for (int st = 128; st > 0; st >>= 1) {
    if (tid < st) { ssum[tid] += ssum[tid + st]; ssq[tid] += ssq[tid + st]; }
    __syncthreads();
  }
  if (tid == 0) {
    float inv = 1.0f / (float)(BB * NN);
    float mean = ssum[0] * inv;
    float var = ssq[0] * inv - mean * mean;
    meanbuf[c] = mean;
    istdbuf[c] = rsqrtf(var + 1e-5f);
  }
}

// ---------------- batchnorm apply ----------------
__global__ void bn_apply_kernel(
    const float* __restrict__ x, const float* __restrict__ attout,
    const float* __restrict__ gamma, const float* __restrict__ meanbuf,
    const float* __restrict__ istdbuf, const float* __restrict__ bnw,
    const float* __restrict__ bnb, float* __restrict__ out) {
  int i = blockIdx.x * blockDim.x + threadIdx.x;
  if (i >= BB * CC * NN) return;
  int c = (i / NN) & (CC - 1);
  float y = gamma[0] * x[i] + attout[i];
  out[i] = (y - meanbuf[c]) * istdbuf[c] * bnw[c] + bnb[c];
}

// ---------------- launcher ----------------
extern "C" void kernel_launch(void* const* d_in, const int* in_sizes, int n_in,
                              void* d_out, int out_size, void* d_ws, size_t ws_size,
                              hipStream_t stream) {
  const float* x     = (const float*)d_in[0];
  const float* wq    = (const float*)d_in[1];
  const float* bq    = (const float*)d_in[2];
  const float* wk    = (const float*)d_in[3];
  const float* bk    = (const float*)d_in[4];
  const float* wv    = (const float*)d_in[5];
  const float* bv    = (const float*)d_in[6];
  const float* freqs = (const float*)d_in[7];
  const float* gamma = (const float*)d_in[8];
  const float* bnw   = (const float*)d_in[9];
  const float* bnb   = (const float*)d_in[10];
  float* out = (float*)d_out;

  char* ws = (char*)d_ws;
  const size_t SZ_W   = (size_t)CC * K2 * 2;          // 73728 B each
  const size_t SZ_QKV = (size_t)BB * NP * CC * 2;     // 3342336 B each
  const size_t OFF_WQ  = 0;
  const size_t OFF_WK  = OFF_WQ + SZ_W;
  const size_t OFF_WV  = OFF_WK + SZ_W;
  const size_t OFF_QR  = OFF_WV + SZ_W;
  const size_t OFF_KR  = OFF_QR + SZ_QKV;
  const size_t OFF_VT  = OFF_KR + SZ_QKV;
  const size_t OFF_ATT = OFF_VT + SZ_QKV;
  const size_t OFF_MEAN = OFF_ATT + (size_t)BB * CC * NN * 4;
  const size_t OFF_ISTD = OFF_MEAN + 256;

  __bf16* wqb = (__bf16*)(ws + OFF_WQ);
  __bf16* wkb = (__bf16*)(ws + OFF_WK);
  __bf16* wvb = (__bf16*)(ws + OFF_WV);
  __bf16* qr  = (__bf16*)(ws + OFF_QR);
  __bf16* kr  = (__bf16*)(ws + OFF_KR);
  __bf16* vT  = (__bf16*)(ws + OFF_VT);
  float* attout  = (float*)(ws + OFF_ATT);
  float* meanbuf = (float*)(ws + OFF_MEAN);
  float* istdbuf = (float*)(ws + OFF_ISTD);

  cvt_w_kernel<<<(CC * K2 + 255) / 256, 256, 0, stream>>>(wq, wk, wv, wqb, wkb, wvb);
  pad_zero_kernel<<<(BB * (NP - NN) * CC + 255) / 256, 256, 0, stream>>>(qr, kr, vT);
  conv_qkv_kernel<<<BB * MT, 96, 0, stream>>>(x, wqb, wkb, wvb, bq, bk, bv, freqs,
                                              qr, kr, vT);
  attn_kernel<<<BB * 51, 128, 0, stream>>>(qr, kr, vT, attout);
  bn_reduce_kernel<<<CC, 256, 0, stream>>>(x, attout, gamma, meanbuf, istdbuf);
  bn_apply_kernel<<<(BB * CC * NN + 255) / 256, 256, 0, stream>>>(
      x, attout, gamma, meanbuf, istdbuf, bnw, bnb, out);
}
